// Attention_87737591923407
// MI455X (gfx1250) — matrix operand order repaired and checked by reading, not compiled, not run on hardware
//
#include <hip/hip_runtime.h>

typedef __attribute__((ext_vector_type(16))) __bf16 v16bf;
typedef __attribute__((ext_vector_type(8)))  float  v8f;

#define WMMA_BF16(a, b, c) \
  __builtin_amdgcn_wmma_f32_16x16x32_bf16(false, (a), false, (b), (short)0, (c), false, false)

// ---------------------------------------------------------------- constants
static constexpr int   Bz   = 2;
static constexpr int   Nq   = 9216;
static constexpr int   Cc   = 512;
static constexpr int   Hh   = 8;
static constexpr int   Dd   = 64;
static constexpr int   NKV  = 576;     // (96/4)*(96/4)
static constexpr int   KCV  = 8192;    // C * 4 * 4 (im2col K)
static constexpr int   Rr   = 32;
static constexpr float SCAL = 4.0f / 32.0f;
static constexpr float QSC  = 0.125f;  // d^-0.5

// ---------------------------------------------------------------- helpers
__device__ __forceinline__ unsigned short f2bf(float f) {
  union { float f; unsigned int u; } v; v.f = f;
  unsigned int x = v.u;
  unsigned int r = (x + 0x7FFFu + ((x >> 16) & 1u)) >> 16;
  return (unsigned short)r;
}

// A fragment (16x32 bf16), ISA 7.12.2: lane = M (0..15), lanes>=16 get K+8.
// element e<8 -> K = k0+khi+e ; e>=8 -> K = k0+16+khi+(e-8). Two 16B loads.
__device__ __forceinline__ v16bf load_a_frag(const unsigned short* A, int ld, int k0) {
  int lane = threadIdx.x & 31;
  int m    = lane & 15;
  int khi  = (lane >> 4) * 8;
  const uint4* p = (const uint4*)(A + (size_t)m * ld + k0 + khi);
  union { v16bf v; uint4 u[2]; } t;
  t.u[0] = p[0];   // 8 bf16: K = k0+khi .. +7
  t.u[1] = p[2];   // 8 bf16: K = k0+khi+16 .. +23
  return t.v;
}

// B fragment (32x16 bf16) where B[k][n] = W[n][k] (W row-major, ld elems).
// col n=lane%16; e<8 -> K=kb+e, e>=8 -> K=16+kb+(e-8), kb=(lane/16)*8. Two 16B loads.
__device__ __forceinline__ v16bf load_b_frag_nk(const unsigned short* W, int ld, int k0) {
  int lane = threadIdx.x & 31;
  int n    = lane & 15;
  int kb   = (lane >> 4) * 8;
  const uint4* p = (const uint4*)(W + (size_t)n * ld + k0 + kb);
  union { v16bf v; uint4 u[2]; } t;
  t.u[0] = p[0];
  t.u[1] = p[2];
  return t.v;
}

// ---------------------------------------------------------------- generic GEMM
// out[M x N] = A[M x K](bf16) * W[N x K]^T(bf16) + bias ; fp32 accum via WMMA.
// grid = (M/64, N/64), block = 128 (4 waves); wave w -> rows m0..m0+15.
// k-loop unrolled 2x with ping-pong fragment sets: loads for step i+1 issue
// while step i's WMMAs execute, and no rotation movs are needed.
// Requires K % 64 == 0 (all call sites: 512, 1024-col weights with K=512, 8192).
__global__ __launch_bounds__(128) void gemm_bf16_kernel(
    const unsigned short* __restrict__ A, const unsigned short* __restrict__ W,
    const float* __restrict__ bias, float* __restrict__ outF,
    unsigned short* __restrict__ outB, int M, int N, int K) {
  int wave = threadIdx.x >> 5;
  int lane = threadIdx.x & 31;
  int m0   = blockIdx.x * 64 + wave * 16;
  int n0   = blockIdx.y * 64;
  const unsigned short* Arow = A + (size_t)m0 * K;
  const unsigned short* Wt[4];
#pragma unroll
  for (int nt = 0; nt < 4; ++nt) Wt[nt] = W + (size_t)(n0 + nt * 16) * K;

  v8f acc[4] = {{}, {}, {}, {}};
  // set A: fragments for k0 = 0
  v16bf aA = load_a_frag(Arow, K, 0);
  v16bf bA[4];
#pragma unroll
  for (int nt = 0; nt < 4; ++nt) bA[nt] = load_b_frag_nk(Wt[nt], K, 0);

  for (int k0 = 0; k0 < K; k0 += 64) {
    // issue set B loads (k0+32), then consume set A
    v16bf aB = load_a_frag(Arow, K, k0 + 32);
    v16bf bB[4];
#pragma unroll
    for (int nt = 0; nt < 4; ++nt) bB[nt] = load_b_frag_nk(Wt[nt], K, k0 + 32);
#pragma unroll
    for (int nt = 0; nt < 4; ++nt) acc[nt] = WMMA_BF16(aA, bA[nt], acc[nt]);
    // issue set A loads (k0+64; dead reload of 0 on last iter), consume set B
    int kn = (k0 + 64 < K) ? k0 + 64 : 0;
    aA = load_a_frag(Arow, K, kn);
#pragma unroll
    for (int nt = 0; nt < 4; ++nt) bA[nt] = load_b_frag_nk(Wt[nt], K, kn);
#pragma unroll
    for (int nt = 0; nt < 4; ++nt) acc[nt] = WMMA_BF16(aB, bB[nt], acc[nt]);
  }

  int mo = (lane >> 4) * 8;
  int nn = lane & 15;
#pragma unroll
  for (int nt = 0; nt < 4; ++nt) {
    int col  = n0 + nt * 16 + nn;
    float bv = bias ? bias[col] : 0.0f;
#pragma unroll
    for (int r = 0; r < 8; ++r) {
      float  res = acc[nt][r] + bv;
      size_t oi  = (size_t)(m0 + mo + r) * N + col;
      if (outF) outF[oi] = res;
      if (outB) outB[oi] = f2bf(res);
    }
  }
}

// ---------------------------------------------------------------- attention
// grid = (Nq/64, H, B), block = 128. LDS: K tile (576x64 bf16), V^T tile
// (64x576 bf16), per-wave score rows (4 x 16 x 576 f32) -> 288 KB (WGP LDS).
__global__ __launch_bounds__(128) void attn_kernel(
    const unsigned short* __restrict__ qp,   // [B*Nq, 512] (col = h*64+dd)
    const unsigned short* __restrict__ kbf,  // [B,H,576,64]
    const unsigned short* __restrict__ vbf,  // [B,H,576,64]
    unsigned short* __restrict__ ob) {       // [B*Nq, 512]
  extern __shared__ char smem[];
  unsigned short* Ks = (unsigned short*)smem;       // 576*64
  unsigned short* Vt = Ks + NKV * Dd;               // 64*576 (transposed V)
  float* P = (float*)(smem + (size_t)2 * NKV * Dd * 2);  // 4*16*576 f32

  int b = blockIdx.z, h = blockIdx.y;
  int tid = threadIdx.x, wave = tid >> 5, lane = tid & 31;
  int qbase = blockIdx.x * 64;

  const unsigned short* kg = kbf + (size_t)(b * Hh + h) * NKV * Dd;
  const unsigned short* vg = vbf + (size_t)(b * Hh + h) * NKV * Dd;
  for (int i = tid * 8; i < NKV * Dd; i += 128 * 8) {
    *(uint4*)(Ks + i) = *(const uint4*)(kg + i);
    uint4 vv = *(const uint4*)(vg + i);
    const unsigned short* vs = (const unsigned short*)&vv;
    int nk = i >> 6, dd = i & 63;
#pragma unroll
    for (int e = 0; e < 8; ++e) Vt[(size_t)(dd + e) * NKV + nk] = vs[e];
  }
  __syncthreads();

  const unsigned short* qrow =
      qp + ((size_t)b * Nq + qbase + wave * 16) * Cc + h * Dd;
  v16bf qa0 = load_a_frag(qrow, Cc, 0);
  v16bf qa1 = load_a_frag(qrow, Cc, 32);

  float* Pw = P + (size_t)wave * 16 * NKV;
  int mo = (lane >> 4) * 8;
  int nn = lane & 15;

  // scores: S = (Q K^T) * d^-0.5 ; 36 column tiles, 2x unrolled ping-pong
  {
    v16bf b0a = load_b_frag_nk(Ks, Dd, 0);
    v16bf b0b = load_b_frag_nk(Ks, Dd, 32);
    for (int nt = 0; nt < 36; nt += 2) {
      v16bf b1a = load_b_frag_nk(Ks + (size_t)(nt + 1) * 16 * Dd, Dd, 0);
      v16bf b1b = load_b_frag_nk(Ks + (size_t)(nt + 1) * 16 * Dd, Dd, 32);
      v8f acc0 = {};
      acc0 = WMMA_BF16(qa0, b0a, acc0);
      acc0 = WMMA_BF16(qa1, b0b, acc0);
#pragma unroll
      for (int r = 0; r < 8; ++r)
        Pw[(size_t)(mo + r) * NKV + nt * 16 + nn] = acc0[r] * QSC;
      int ntn = (nt + 2 < 36) ? nt + 2 : 0;  // dead reload on last iter
      b0a = load_b_frag_nk(Ks + (size_t)ntn * 16 * Dd, Dd, 0);
      b0b = load_b_frag_nk(Ks + (size_t)ntn * 16 * Dd, Dd, 32);
      v8f acc1 = {};
      acc1 = WMMA_BF16(qa0, b1a, acc1);
      acc1 = WMMA_BF16(qa1, b1b, acc1);
#pragma unroll
      for (int r = 0; r < 8; ++r)
        Pw[(size_t)(mo + r) * NKV + (nt + 1) * 16 + nn] = acc1[r] * QSC;
    }
  }
  __syncthreads();

  // softmax: 2 lanes per row (halves of 288), combine via shfl_xor(16)
  int row = lane & 15, half = lane >> 4;
  float* pr = Pw + (size_t)row * NKV + half * 288;
  float mx = -3.0e38f;
  for (int j = 0; j < 288; ++j) mx = fmaxf(mx, pr[j]);
  mx = fmaxf(mx, __shfl_xor(mx, 16, 32));
  float s = 0.0f;
  for (int j = 0; j < 288; ++j) { float e = __expf(pr[j] - mx); pr[j] = e; s += e; }
  s += __shfl_xor(s, 16, 32);
  float inv = 1.0f / s;
  __syncthreads();
  if (half == 0) {  // in-place f32 -> bf16 repack (write byte 2j < read byte 4j)
    float* pf = Pw + (size_t)row * NKV;
    unsigned short* pb = (unsigned short*)pf;
    for (int j = 0; j < NKV; ++j) pb[j] = f2bf(pf[j] * inv);
  }
  __syncthreads();

  // O = P V : A = P (bf16, row pitch 1152 ushorts), B from transposed V.
  // 18 k-steps, 2x unrolled ping-pong.
  const unsigned short* Pb = (const unsigned short*)Pw;  // ld = 576*2 = 1152
  size_t orow = ((size_t)b * Nq + qbase + wave * 16) * Cc + h * Dd;
#pragma unroll
  for (int dt = 0; dt < 4; ++dt) {
    int dd0 = dt * 16;
    const unsigned short* Vtd = Vt + (size_t)dd0 * NKV;
    v8f acc = {};
    v16bf aA = load_a_frag(Pb, 2 * NKV, 0);
    v16bf vA = load_b_frag_nk(Vtd, NKV, 0);
    for (int k0 = 0; k0 < NKV; k0 += 64) {
      v16bf aB = load_a_frag(Pb, 2 * NKV, k0 + 32);
      v16bf vB = load_b_frag_nk(Vtd, NKV, k0 + 32);
      acc = WMMA_BF16(aA, vA, acc);
      int kn = (k0 + 64 < NKV) ? k0 + 64 : 0;  // dead reload on last iter
      aA = load_a_frag(Pb, 2 * NKV, kn);
      vA = load_b_frag_nk(Vtd, NKV, kn);
      acc = WMMA_BF16(aB, vB, acc);
    }
#pragma unroll
    for (int r = 0; r < 8; ++r)
      ob[orow + (size_t)(mo + r) * Cc + dd0 + nn] = f2bf(acc[r]);
  }
}

// ---------------------------------------------------------------- small kernels
__global__ __launch_bounds__(256) void cvt_f2bf_kernel(
    const float* __restrict__ in, unsigned short* __restrict__ out, int n) {
  int i = blockIdx.x * 256 + threadIdx.x;
  if (i < n) out[i] = f2bf(in[i]);
}

// w_eff = w_q + s * (B_q @ A_q)   (LoRA folded into the weight)
__global__ __launch_bounds__(256) void wq_eff_kernel(
    const float* __restrict__ wq, const float* __restrict__ Aq,
    const float* __restrict__ Bq, unsigned short* __restrict__ out) {
  int i = blockIdx.x * 256 + threadIdx.x;  // 512*512
  int o = i >> 9, c = i & 511;
  float acc = wq[i];
#pragma unroll
  for (int r = 0; r < Rr; ++r) acc += SCAL * Bq[o * Rr + r] * Aq[r * Cc + c];
  out[i] = f2bf(acc);
}

// w_lv = s * (B_v @ A_v)
__global__ __launch_bounds__(256) void wlv_kernel(
    const float* __restrict__ Av, const float* __restrict__ Bv,
    unsigned short* __restrict__ out) {
  int i = blockIdx.x * 256 + threadIdx.x;  // 512*512
  int o = i >> 9, c = i & 511;
  float acc = 0.0f;
#pragma unroll
  for (int r = 0; r < Rr; ++r) acc += SCAL * Bv[o * Rr + r] * Av[r * Cc + c];
  out[i] = f2bf(acc);
}

// wcol[o][p*512+i] = w_sr[o][i][ph][pw]  (OIHW -> [512 x 8192] GEMM weight)
__global__ __launch_bounds__(256) void wcol_kernel(
    const float* __restrict__ wsr, unsigned short* __restrict__ out) {
  int i = blockIdx.x * 256 + threadIdx.x;  // 512*8192
  int o = i >> 13, col = i & 8191;
  int p = col >> 9, ci = col & 511;
  out[i] = f2bf(wsr[(size_t)(o * Cc + ci) * 16 + p]);
}

// im2col: xcol[(b,oh,ow)][p*512+i] = x[b][(oh*4+ph)*96 + ow*4+pw][i]
__global__ __launch_bounds__(256) void xcol_kernel(
    const float* __restrict__ x, unsigned short* __restrict__ out) {
  int i = blockIdx.x * 256 + threadIdx.x;  // 1152*8192
  int col = i & 8191, rowkv = i >> 13;
  int b = rowkv / NKV, nk = rowkv % NKV;
  int oh = nk / 24, ow = nk % 24;
  int p = col >> 9, ci = col & 511;
  int ph = p >> 2, pw = p & 3;
  int pix = (oh * 4 + ph) * 96 + ow * 4 + pw;
  out[i] = f2bf(x[((size_t)b * Nq + pix) * Cc + ci]);
}

// LayerNorm over C=512; one block per row, fp32 in -> bf16 out
__global__ __launch_bounds__(128) void ln_kernel(
    const float* __restrict__ xin, const float* __restrict__ g,
    const float* __restrict__ bb, unsigned short* __restrict__ outbf) {
  __shared__ float s1[128], s2[128];
  int row = blockIdx.x, t = threadIdx.x;
  const float* xr = xin + (size_t)row * Cc;
  float a0 = xr[t], a1 = xr[t + 128], a2 = xr[t + 256], a3 = xr[t + 384];
  s1[t] = a0 + a1 + a2 + a3;
  s2[t] = a0 * a0 + a1 * a1 + a2 * a2 + a3 * a3;
  __syncthreads();
  for (int off = 64; off > 0; off >>= 1) {
    if (t < off) { s1[t] += s1[t + off]; s2[t] += s2[t + off]; }
    __syncthreads();
  }
  float mu   = s1[0] * (1.0f / Cc);
  float var  = s2[0] * (1.0f / Cc) - mu * mu;
  float rinv = rsqrtf(var + 1e-5f);
  unsigned short* orow = outbf + (size_t)row * Cc;
  orow[t]       = f2bf((a0 - mu) * rinv * g[t]       + bb[t]);
  orow[t + 128] = f2bf((a1 - mu) * rinv * g[t + 128] + bb[t + 128]);
  orow[t + 256] = f2bf((a2 - mu) * rinv * g[t + 256] + bb[t + 256]);
  orow[t + 384] = f2bf((a3 - mu) * rinv * g[t + 384] + bb[t + 384]);
}

// split kv -> k,v bf16 (B,H,576,64); add lv with the reference's raw reshape
__global__ __launch_bounds__(256) void packkv_kernel(
    const float* __restrict__ kvf, const float* __restrict__ lvf,
    unsigned short* __restrict__ kb, unsigned short* __restrict__ vb) {
  int i = blockIdx.x * 256 + threadIdx.x;  // B*H*576*64
  int dd = i & 63;
  int t  = i >> 6;
  int m  = t % NKV;
  int h  = (t / NKV) % Hh;
  int b  = t / (NKV * Hh);
  size_t kvrow = (size_t)(b * NKV + m) * (2 * Cc);
  kb[i] = f2bf(kvf[kvrow + h * Dd + dd]);
  float vv = kvf[kvrow + Cc + h * Dd + dd] +
             lvf[(size_t)b * NKV * Cc + (size_t)h * NKV * Dd + m * Dd + dd];
  vb[i] = f2bf(vv);
}

// ---------------------------------------------------------------- launch
extern "C" void kernel_launch(void* const* d_in, const int* in_sizes, int n_in,
                              void* d_out, int out_size, void* d_ws, size_t ws_size,
                              hipStream_t stream) {
  const float* x      = (const float*)d_in[0];
  const float* w_q    = (const float*)d_in[1];
  const float* b_q    = (const float*)d_in[2];
  const float* w_kv   = (const float*)d_in[3];
  const float* b_kv   = (const float*)d_in[4];
  const float* w_proj = (const float*)d_in[5];
  const float* b_proj = (const float*)d_in[6];
  const float* w_sr   = (const float*)d_in[7];
  const float* b_sr   = (const float*)d_in[8];
  const float* ln_g   = (const float*)d_in[9];
  const float* ln_b   = (const float*)d_in[10];
  const float* lAq    = (const float*)d_in[11];
  const float* lBq    = (const float*)d_in[12];
  const float* lAv    = (const float*)d_in[13];
  const float* lBv    = (const float*)d_in[14];
  (void)in_sizes; (void)n_in; (void)out_size;

  char* ws = (char*)d_ws; (void)ws_size;
  size_t off = 0;
  auto alloc = [&](size_t bytes) { size_t o = off; off += (bytes + 255) & ~(size_t)255; return o; };
  const size_t o_xb    = alloc((size_t)Bz * Nq * Cc * 2);    // x bf16 (later aliased as attn out)
  const size_t o_xcol  = alloc((size_t)Bz * NKV * KCV * 2);  // im2col bf16
  const size_t o_qpl   = alloc((size_t)Bz * Nq * Cc * 2);    // q bf16
  const size_t o_wq    = alloc((size_t)Cc * Cc * 2);
  const size_t o_wlv   = alloc((size_t)Cc * Cc * 2);
  const size_t o_wkv   = alloc((size_t)2 * Cc * Cc * 2);
  const size_t o_wpr   = alloc((size_t)Cc * Cc * 2);
  const size_t o_wcol  = alloc((size_t)Cc * KCV * 2);
  const size_t o_xsr   = alloc((size_t)Bz * NKV * Cc * 4);   // conv out f32
  const size_t o_xsbf  = alloc((size_t)Bz * NKV * Cc * 2);   // post-LN bf16
  const size_t o_kvf   = alloc((size_t)Bz * NKV * 2 * Cc * 4);
  const size_t o_lvf   = alloc((size_t)Bz * NKV * Cc * 4);
  const size_t o_kbf   = alloc((size_t)Bz * Hh * NKV * Dd * 2);
  const size_t o_vbf   = alloc((size_t)Bz * Hh * NKV * Dd * 2);
  auto U = [&](size_t o) { return (unsigned short*)(ws + o); };
  auto F = [&](size_t o) { return (float*)(ws + o); };

  // --- weight prep / conversion ---
  wq_eff_kernel<<<Cc * Cc / 256, 256, 0, stream>>>(w_q, lAq, lBq, U(o_wq));
  wlv_kernel<<<Cc * Cc / 256, 256, 0, stream>>>(lAv, lBv, U(o_wlv));
  cvt_f2bf_kernel<<<2 * Cc * Cc / 256, 256, 0, stream>>>(w_kv, U(o_wkv), 2 * Cc * Cc);
  cvt_f2bf_kernel<<<Cc * Cc / 256, 256, 0, stream>>>(w_proj, U(o_wpr), Cc * Cc);
  wcol_kernel<<<Cc * KCV / 256, 256, 0, stream>>>(w_sr, U(o_wcol));

  // --- KV path: im2col -> conv GEMM -> LN -> kv/lv GEMMs -> pack ---
  xcol_kernel<<<Bz * NKV * KCV / 256, 256, 0, stream>>>(x, U(o_xcol));
  gemm_bf16_kernel<<<dim3(Bz * NKV / 64, Cc / 64), 128, 0, stream>>>(
      U(o_xcol), U(o_wcol), b_sr, F(o_xsr), nullptr, Bz * NKV, Cc, KCV);
  ln_kernel<<<Bz * NKV, 128, 0, stream>>>(F(o_xsr), ln_g, ln_b, U(o_xsbf));
  gemm_bf16_kernel<<<dim3(Bz * NKV / 64, 2 * Cc / 64), 128, 0, stream>>>(
      U(o_xsbf), U(o_wkv), b_kv, F(o_kvf), nullptr, Bz * NKV, 2 * Cc, Cc);
  gemm_bf16_kernel<<<dim3(Bz * NKV / 64, Cc / 64), 128, 0, stream>>>(
      U(o_xsbf), U(o_wlv), nullptr, F(o_lvf), nullptr, Bz * NKV, Cc, Cc);
  packkv_kernel<<<Bz * Hh * NKV * Dd / 256, 256, 0, stream>>>(
      F(o_kvf), F(o_lvf), U(o_kbf), U(o_vbf));

  // --- Q path (LoRA folded into weight) ---
  cvt_f2bf_kernel<<<Bz * Nq * Cc / 256, 256, 0, stream>>>(x, U(o_xb), Bz * Nq * Cc);
  gemm_bf16_kernel<<<dim3(Bz * Nq / 64, Cc / 64), 128, 0, stream>>>(
      U(o_xb), U(o_wq), b_q, nullptr, U(o_qpl), Bz * Nq, Cc, Cc);

  // --- attention (K, V^T and scores fully LDS-resident: 288 KB / WGP) ---
  const int attn_lds = NKV * Dd * 2 * 2 + 4 * 16 * NKV * 4;  // 294912 B
  hipFuncSetAttribute((const void*)attn_kernel,
                      hipFuncAttributeMaxDynamicSharedMemorySize, attn_lds);
  attn_kernel<<<dim3(Nq / 64, Hh, Bz), 128, attn_lds, stream>>>(
      U(o_qpl), U(o_kbf), U(o_vbf), U(o_xb));  // output aliases dead x-bf16 buffer

  // --- output projection (fp32 out) ---
  gemm_bf16_kernel<<<dim3(Bz * Nq / 64, Cc / 64), 128, 0, stream>>>(
      U(o_xb), U(o_wpr), b_proj, (float*)d_out, nullptr, Bz * Nq, Cc, Cc);
}
